// EnsembleFC_72387378806934
// MI455X (gfx1250) — matrix operand order repaired and checked by reading, not compiled, not run on hardware
//
#include <hip/hip_runtime.h>

typedef __bf16 bf16;
typedef bf16  v16bf __attribute__((ext_vector_type(16)));
typedef float v8f   __attribute__((ext_vector_type(8)));
typedef unsigned int u32x4 __attribute__((ext_vector_type(4)));
typedef unsigned int u32x2 __attribute__((ext_vector_type(2)));
typedef int   i32x8 __attribute__((ext_vector_type(8)));
typedef int   i32x4 __attribute__((ext_vector_type(4)));
typedef float f32x4 __attribute__((ext_vector_type(4)));

#define AS3 __attribute__((address_space(3)))

constexpr int E     = 16;
constexpr int DIN   = 512;
constexpr int H     = 512;
constexpr int BATCH = 8192;
constexpr int BM    = 64;        // batch rows per workgroup
constexpr int SA    = DIN + 8;   // LDS row stride (bf16) for activation buffers
constexpr int SW    = 32 + 8;    // LDS row stride (bf16) for weight k-slice

// Dynamic-LDS byte offsets (no static LDS in the fused kernel -> base 0;
// the TDM descriptors and the AS3 offset addressing both rely on this).
constexpr unsigned OFF_A  = 0;
constexpr unsigned OFF_B  = BM * SA * 2;                 // 66560
constexpr unsigned OFF_W0 = 2 * BM * SA * 2;             // 133120
constexpr unsigned WSLICE = 512 * SW * 2;                // 40960
constexpr unsigned OFF_W1 = OFF_W0 + WSLICE;             // 174080
constexpr unsigned SMEM_BYTES = OFF_W1 + WSLICE;         // 215040

union FragU { v16bf v; u32x4 q[2]; };

// 32-bit LDS accessors (inttoptr to addrspace(3): no generic null checks,
// no 64-bit address math; constant parts fold into ds offset immediates).
__device__ inline u32x4 lds_read_v4(unsigned off) { return *(AS3 const u32x4*)off; }
__device__ inline void  lds_write_v4(unsigned off, u32x4 v) { *(AS3 u32x4*)off = v; }
__device__ inline void  lds_write_h(unsigned off, bf16 v)   { *(AS3 bf16*)off = v; }
__device__ inline float lds_read_hf(unsigned off) { return (float)*(AS3 const bf16*)off; }
__device__ inline void  lds_write_f(unsigned off, float v)  { *(AS3 float*)off = v; }
__device__ inline float lds_read_f(unsigned off)  { return *(AS3 const float*)off; }

#define HAVE_TDM __has_builtin(__builtin_amdgcn_tensor_load_to_lds)

// ---------------------------------------------------------------------------
// Tensor Data Mover: 2-D tile load (global -> LDS) with LDS row padding.
// 6-arg builtin: (g0:u32x4, g1:i32x8, g2:i32x4, g3:i32x4, i32x8, cpol)
// ---------------------------------------------------------------------------
#if HAVE_TDM
__device__ inline void tdm_load_2d(unsigned ldsAddr, const void* gptr,
                                   unsigned tensorD0, unsigned tensorD1,
                                   unsigned tileD0, unsigned tileD1,
                                   unsigned strideD0,
                                   unsigned padInterval, unsigned padAmount,
                                   bool padEn) {
  const unsigned long long ga = (unsigned long long)gptr;
  u32x4 g0 = { 1u,                                   // count=1, user mode
               ldsAddr,                               // lds_addr [63:32]
               (unsigned)ga,                          // global_addr lo
               (unsigned)((ga >> 32) & 0x1FFFFFFull)  // global_addr [56:32]
                   | (2u << 30) };                    // type=2 (image)
  const unsigned w0 = (1u << 16)                      // data_size = 2 bytes
                    | (padEn ? ((1u << 20) | (padInterval << 22)
                                | (padAmount << 25)) : 0u);
  i32x8 g1 = { (int)w0,
               (int)(tensorD0 << 16),                          // dim0[15:0]
               (int)((tensorD0 >> 16) | (tensorD1 << 16)),     // dim0 hi | dim1 lo
               (int)((tensorD1 >> 16) | (tileD0 << 16)),       // dim1 hi | tile0
               (int)(tileD1 & 0xFFFFu),                        // tile1 (tile2=0)
               (int)strideD0,                                  // dim0_stride lo32
               0, 0 };
  i32x4 gz4 = { 0, 0, 0, 0 };
  i32x8 gz8 = { 0, 0, 0, 0, 0, 0, 0, 0 };
  __builtin_amdgcn_tensor_load_to_lds(g0, g1, gz4, gz4, gz8, 0);
}
#endif

// Fallback: cooperative global->LDS copy of one 512x32 weight slice.
__device__ inline void stage_w_slice_coop(unsigned wOff, const bf16* gsrc, int tid) {
  #pragma unroll
  for (int i = 0; i < 8; ++i) {
    const int idx = tid * 8 + i;     // 0..2047 u32x4 chunks (4 per column)
    const int col = idx >> 2;
    const int c4  = idx & 3;
    lds_write_v4(wOff + (unsigned)(col * SW + c4 * 8) * 2,
                 *(const u32x4*)(gsrc + (size_t)col * 512 + c4 * 8));
  }
}

// A fragment (16x32 bf16, M x K), contiguous k order: lane l -> m=l&15;
// group g=l>>4: elems 0..15 = K(g*16..+15).
__device__ inline v16bf load_a_frag(unsigned aOff, int row0, int k0, int lane) {
  const int m = lane & 15, g = lane >> 4;
  const unsigned p = aOff + (unsigned)((row0 + m) * SA + k0 + g * 16) * 2;
  FragU f;
  f.q[0] = lds_read_v4(p);
  f.q[1] = lds_read_v4(p + 16);    // +8 elements
  return f.v;
}

// ---------------------------------------------------------------------------
// One fused GEMM+bias+ReLU layer: [BM x 512](LDS @ aOff) x [512 x 512](Wt,
// bf16, transposed [N][K] in global) -> bf16 [BM x 512] into LDS @ outOff.
// Weight k-slices double-buffered via TDM (OFF_W0/OFF_W1).
// ---------------------------------------------------------------------------
__device__ inline void gemm_relu_layer(unsigned aOff, unsigned outOff,
                                       const bf16* WtE,
                                       const float* biasE, int tid) {
  const int lane = tid & 31;
  const int wave = tid >> 5;
  const int rs = wave >> 1;   // row strip 0..3 (16 rows)
  const int ch = wave & 1;    // column half (256 cols)

  v8f acc[16] = {};           // 16 tiles of 16x16 f32

  // Prologue: slice 0 -> W0 (also flushes any prior outstanding TDM).
#if HAVE_TDM
  if (wave == 0) {
    tdm_load_2d(OFF_W0, WtE, 512, 512, 32, 512, 512, 3, 3, true);
    __builtin_amdgcn_s_wait_tensorcnt(0);
  }
#else
  stage_w_slice_coop(OFF_W0, WtE, tid);
#endif
  __syncthreads();

  // Per-wave B base (lane n -> column, group g -> K half); column tiles are
  // constant +1280B steps -> ds offset immediates.
  const unsigned bLane = (unsigned)((ch * 256 + (lane & 15)) * SW + (lane >> 4) * 16) * 2;

  for (int ks = 0; ks < 16; ++ks) {        // K = 512 in steps of 32
    const unsigned wOff = OFF_W0 + (unsigned)(ks & 1) * WSLICE;

    // Kick DMA for slice ks+1 into the other buffer (its readers finished
    // before the barrier that ended iteration ks-1).
#if HAVE_TDM
    if (wave == 0 && ks < 15) {
      tdm_load_2d(OFF_W0 + (unsigned)(~ks & 1) * WSLICE,
                  WtE + (size_t)(ks + 1) * 32, 512, 512, 32, 512, 512, 3, 3, true);
    }
#else
    if (ks < 15)
      stage_w_slice_coop(OFF_W0 + (unsigned)(~ks & 1) * WSLICE,
                         WtE + (size_t)(ks + 1) * 32, tid);
#endif

    const v16bf a = load_a_frag(aOff, rs * 16, ks * 32, lane);
    const unsigned bbase = wOff + bLane;

    // Software-pipelined: load B frag nt+1 while wmma consumes frag nt.
    FragU fcur, fnxt;
    fcur.q[0] = lds_read_v4(bbase);
    fcur.q[1] = lds_read_v4(bbase + 16);
    #pragma unroll
    for (int nt = 0; nt < 16; ++nt) {
      if (nt < 15) {
        fnxt.q[0] = lds_read_v4(bbase + (unsigned)(nt + 1) * (16 * SW * 2));
        fnxt.q[1] = lds_read_v4(bbase + (unsigned)(nt + 1) * (16 * SW * 2) + 16);
      }
      acc[nt] = __builtin_amdgcn_wmma_f32_16x16x32_bf16(
          false, a, false, fcur.v, (short)0, acc[nt], false, false);
      fcur = fnxt;
    }

#if HAVE_TDM
    if (wave == 0 && ks < 15) __builtin_amdgcn_s_wait_tensorcnt(0);
#endif
    __syncthreads();   // publishes next slice; retires reads of current one
  }

  // Epilogue: bias + ReLU, pack to bf16.
  // C layout: elem r at lane l -> row = r + (l>>4)*8, col = l&15.
  #pragma unroll
  for (int nt = 0; nt < 16; ++nt) {
    const int col  = ch * 256 + nt * 16 + (lane & 15);
    const float bs = biasE[col];
    #pragma unroll
    for (int r = 0; r < 8; ++r) {
      const int row = rs * 16 + (lane >> 4) * 8 + r;
      float v = acc[nt][r] + bs;
      v = v > 0.f ? v : 0.f;
      lds_write_h(outOff + (unsigned)(row * SA + col) * 2, (bf16)v);
    }
  }
  __syncthreads();
}

__global__ __launch_bounds__(256) void ensemble_mlp_fused(
    const bf16* __restrict__ xbf,
    const bf16* __restrict__ W1t, const float* __restrict__ b1,
    const bf16* __restrict__ W2t, const float* __restrict__ b2,
    const float* __restrict__ W3, const float* __restrict__ b3,
    float* __restrict__ out) {
  const int tid  = threadIdx.x;
  const int wave = tid >> 5;
  const int e    = blockIdx.y;
  const int b0   = blockIdx.x * BM;

  // Stage x tile (64 x 512 bf16) into LDS @ OFF_A.
  const bf16* xsrc = xbf + (size_t)b0 * DIN;
#if HAVE_TDM
  if (wave == 0)   // waited inside layer-1 prologue (TENSORcnt is in-order)
    tdm_load_2d(OFF_A, xsrc, 512, 64, 512, 64, 512, 7, 3, true);
#else
  #pragma unroll
  for (int i = 0; i < 16; ++i) {
    const int idx = tid + i * 256;
    const int row = idx >> 6, c = idx & 63;
    lds_write_v4(OFF_A + (unsigned)(row * SA + c * 8) * 2,
                 *(const u32x4*)(xsrc + row * DIN + c * 8));
  }
#endif

  gemm_relu_layer(OFF_A, OFF_B, W1t + (size_t)e * DIN * H, b1 + e * H, tid);
  gemm_relu_layer(OFF_B, OFF_A, W2t + (size_t)e * H * H,   b2 + e * H, tid);

  // Layer 3: out[e,b] = dot(h2[b,:], W3[e,:,0]) + b3[e]; 4 threads per row.
  {
    const int row = tid >> 2, part = tid & 3;
    const float* w3 = W3 + (size_t)e * H;
    const unsigned h2o = OFF_A + (unsigned)(row * SA + part * 128) * 2;
    float s = 0.f;
    #pragma unroll 8
    for (int k = 0; k < 128; ++k) s += lds_read_hf(h2o + (unsigned)k * 2) * w3[part * 128 + k];
    lds_write_f(OFF_B + (unsigned)tid * 4, s);
    __syncthreads();
    if (part == 0) {
      const unsigned ro = OFF_B + (unsigned)tid * 4;
      const float r = lds_read_f(ro) + lds_read_f(ro + 4) + lds_read_f(ro + 8) +
                      lds_read_f(ro + 12) + b3[e];
      out[(size_t)e * BATCH + b0 + row] = r;
    }
  }
}

// fp32 -> bf16 (vectorized x4)
__global__ void cvt_f32_bf16(const float* __restrict__ in, bf16* __restrict__ out, int n4) {
  const int i = blockIdx.x * blockDim.x + threadIdx.x;
  if (i >= n4) return;
  const f32x4 v = *(const f32x4*)(in + (size_t)i * 4);
  union { bf16 r[4]; u32x2 q; } u;
  u.r[0] = (bf16)v.x; u.r[1] = (bf16)v.y; u.r[2] = (bf16)v.z; u.r[3] = (bf16)v.w;
  *(u32x2*)(out + (size_t)i * 4) = u.q;
}

// W[e][k][n] fp32 -> Wt[e][n][k] bf16, 32x32 LDS-tiled transpose
__global__ void transpose_w_to_bf16(const float* __restrict__ in, bf16* __restrict__ out) {
  __shared__ float tile[32][33];
  const int e  = blockIdx.z;
  const int kb = blockIdx.x * 32, nb = blockIdx.y * 32;
  const float* src = in + (size_t)e * 512 * 512;
  bf16* dst = out + (size_t)e * 512 * 512;
  #pragma unroll
  for (int r = 0; r < 32; r += 8)
    tile[threadIdx.y + r][threadIdx.x] =
        src[(size_t)(kb + threadIdx.y + r) * 512 + nb + threadIdx.x];
  __syncthreads();
  #pragma unroll
  for (int r = 0; r < 32; r += 8)
    dst[(size_t)(nb + threadIdx.y + r) * 512 + kb + threadIdx.x] =
        (bf16)tile[threadIdx.x][threadIdx.y + r];
}

extern "C" void kernel_launch(void* const* d_in, const int* in_sizes, int n_in,
                              void* d_out, int out_size, void* d_ws, size_t ws_size,
                              hipStream_t stream) {
  const float* x  = (const float*)d_in[0];
  const float* W1 = (const float*)d_in[1];
  const float* b1 = (const float*)d_in[2];
  const float* W2 = (const float*)d_in[3];
  const float* b2 = (const float*)d_in[4];
  const float* W3 = (const float*)d_in[5];
  const float* b3 = (const float*)d_in[6];
  float* out = (float*)d_out;

  // workspace: xbf (8 MB) | W1t (8 MB) | W2t (8 MB)
  bf16* xbf = (bf16*)d_ws;
  bf16* W1t = xbf + (size_t)BATCH * DIN;
  bf16* W2t = W1t + (size_t)E * DIN * H;

  const int n4 = BATCH * DIN / 4;
  cvt_f32_bf16<<<(n4 + 255) / 256, 256, 0, stream>>>(x, xbf, n4);

  dim3 tb(32, 8), tg(512 / 32, 512 / 32, E);
  transpose_w_to_bf16<<<tg, tb, 0, stream>>>(W1, W1t);
  transpose_w_to_bf16<<<tg, tb, 0, stream>>>(W2, W2t);

  dim3 grid(BATCH / BM, E);
  ensemble_mlp_fused<<<grid, 256, SMEM_BYTES, stream>>>(xbf, W1t, b1, W2t, b2, W3, b3, out);
}
